// AdaptivePoolAttention_84370337562731
// MI455X (gfx1250) — matrix operand order repaired and checked by reading, not compiled, not run on hardware
//
#include <hip/hip_runtime.h>
#include <cstdint>
#include <cstddef>

typedef __attribute__((ext_vector_type(16))) __bf16 v16bf;
typedef __attribute__((ext_vector_type(8)))  float  v8f;

// ---------------------------------------------------------------------------
// WMMA helpers (CDNA5 wave32, v_wmma_f32_16x16x32_bf16)
// ---------------------------------------------------------------------------
__device__ __forceinline__ v8f wmma_bf16(v16bf a, v16bf b, v8f c) {
    return __builtin_amdgcn_wmma_f32_16x16x32_bf16(
        /*neg_a=*/false, a, /*neg_b=*/false, b,
        /*c_mod=*/(short)0, c, /*reuse_a=*/false, /*reuse_b=*/false);
}

// A fragment: 16x32 bf16, source row-major [m][k], ld in elements.
__device__ __forceinline__ v16bf frag_a(const __bf16* p, int ld, int lane) {
    int m  = lane & 15;
    int kh = (lane >> 4) << 4;
    const __bf16* q = p + m * ld + kh;
    v16bf a;
#pragma unroll
    for (int j = 0; j < 16; ++j) a[j] = q[j];
    return a;
}

// B fragment: 32x16 bf16, source stored [n][k].
__device__ __forceinline__ v16bf frag_b(const __bf16* p, int ld, int lane) {
    int n  = lane & 15;
    int kh = (lane >> 4) << 4;
    const __bf16* q = p + n * ld + kh;
    v16bf b;
#pragma unroll
    for (int j = 0; j < 16; ++j) b[j] = q[j];
    return b;
}

// ---------------------------------------------------------------------------
// CDNA5 async global->LDS copy (ASYNCcnt path, gfx1250)
// ---------------------------------------------------------------------------
__device__ __forceinline__ unsigned lds_off(const void* p) {
    return (unsigned)(uintptr_t)p;   // LDS aperture: low 32 bits are the LDS byte offset
}
__device__ __forceinline__ void async_copy_b128(unsigned lds_byte, const void* gaddr) {
    asm volatile("global_load_async_to_lds_b128 %0, %1, off"
                 :: "v"(lds_byte), "v"(gaddr) : "memory");
}
#define WAIT_ASYNC(N) asm volatile("s_wait_asynccnt " #N ::: "memory")

// ---------------------------------------------------------------------------
// Layout-prep kernels
// ---------------------------------------------------------------------------
__global__ void k_f32_to_bf16(const float* __restrict__ in, __bf16* __restrict__ out, size_t n) {
    size_t i = (size_t)blockIdx.x * blockDim.x + threadIdx.x;
    if (i < n) out[i] = (__bf16)in[i];
}

__global__ void k_transpose_bf16(const float* __restrict__ in, __bf16* __restrict__ out,
                                 int R, int Cc) {
    size_t i = (size_t)blockIdx.x * blockDim.x + threadIdx.x;
    if (i < (size_t)R * Cc) {
        int r = (int)(i / Cc), c = (int)(i % Cc);
        out[(size_t)c * R + r] = (__bf16)in[i];
    }
}

__global__ void k_rearrange_wp(const float* __restrict__ in, __bf16* __restrict__ out) {
    int i = blockIdx.x * blockDim.x + threadIdx.x;
    if (i < 64 * 3136) {
        int d = i / 3136, rem = i % 3136;
        int hw = rem >> 6, c = rem & 63;
        out[i] = (__bf16)in[(d * 64 + c) * 49 + hw];
    }
}

__global__ void k_relpos(const float* __restrict__ in, __bf16* __restrict__ out) {
    int i = blockIdx.x * blockDim.x + threadIdx.x;
    if (i < 128 * 64) out[i] = (i >> 6) < 127 ? (__bf16)in[i] : (__bf16)0.0f;
}

// ---------------------------------------------------------------------------
// Stage 1: qkv = x @ W_qkv (M=50176, K=768, N=2304).
// 128x64 workgroup tile, 8 waves each 32x32 (4 accumulators, 8 WMMA / k-tile),
// k-tile 64, double-buffered LDS fed by async global->LDS copies.
// Output scattered to blocked bf16 layout qkv2[(sel*12+n)][bt][hw*64+c].
// ---------------------------------------------------------------------------
__global__ void __launch_bounds__(256) k_gemm_qkv(const __bf16* __restrict__ xb,
                                                  const __bf16* __restrict__ wT,
                                                  __bf16* __restrict__ qkv2) {
    __shared__ __bf16 As[2][128][72];
    __shared__ __bf16 Bs[2][64][72];
    int tid = threadIdx.x, wave = tid >> 5, lane = tid & 31;
    int m0 = (wave & 3) * 32;        // wave M offset within 128
    int n0 = (wave >> 2) * 32;       // wave N offset within 64
    int row0 = blockIdx.x * 128, col0 = blockIdx.y * 64;

    // 16B async chunks: A tile 128x64 -> 1024 chunks (4/thread); B 64x64 -> 512 (2/thread)
    auto issue = [&](int buf, int kt) {
#pragma unroll
        for (int q = 0; q < 4; ++q) {
            int cidx = tid * 4 + q;
            int r = cidx >> 3, c8 = (cidx & 7) * 8;
            const __bf16* src = xb + (size_t)(row0 + r) * 768 + kt + c8;
            async_copy_b128(lds_off(&As[buf][r][c8]), src);
        }
#pragma unroll
        for (int q = 0; q < 2; ++q) {
            int cidx = tid * 2 + q;
            int r = cidx >> 3, c8 = (cidx & 7) * 8;
            const __bf16* src = wT + (size_t)(col0 + r) * 768 + kt + c8;
            async_copy_b128(lds_off(&Bs[buf][r][c8]), src);
        }
    };

    v8f acc[2][2];
    v8f zero = {};
#pragma unroll
    for (int i = 0; i < 2; ++i)
#pragma unroll
        for (int j = 0; j < 2; ++j) acc[i][j] = zero;

    issue(0, 0);
    for (int it = 0; it < 12; ++it) {        // 768 / 64
        int cur = it & 1;
        if (it + 1 < 12) {
            issue(cur ^ 1, (it + 1) * 64);
            WAIT_ASYNC(6);                   // buffer `cur` complete (in-order)
        } else {
            WAIT_ASYNC(0);
        }
        __syncthreads();
#pragma unroll
        for (int ks = 0; ks < 64; ks += 32) {
            v16bf a0 = frag_a(&As[cur][m0][ks],      72, lane);
            v16bf a1 = frag_a(&As[cur][m0 + 16][ks], 72, lane);
            v16bf b0 = frag_b(&Bs[cur][n0][ks],      72, lane);
            v16bf b1 = frag_b(&Bs[cur][n0 + 16][ks], 72, lane);
            acc[0][0] = wmma_bf16(a0, b0, acc[0][0]);
            acc[0][1] = wmma_bf16(a0, b1, acc[0][1]);
            acc[1][0] = wmma_bf16(a1, b0, acc[1][0]);
            acc[1][1] = wmma_bf16(a1, b1, acc[1][1]);
        }
        __syncthreads();
    }

    int n = lane & 15, mo = (lane >> 4) * 8;
#pragma unroll
    for (int sm = 0; sm < 2; ++sm)
#pragma unroll
        for (int sn = 0; sn < 2; ++sn)
#pragma unroll
            for (int j = 0; j < 8; ++j) {
                int row = row0 + m0 + sm * 16 + mo + j;
                int col = col0 + n0 + sn * 16 + n;
                int bt = row / 49, hw = row % 49;
                int sel = col / 768, rem = col % 768;
                int nn = rem >> 6, c = rem & 63;
                size_t dst = ((size_t)(sel * 12 + nn) * 1024 + bt) * 3136 + hw * 64 + c;
                qkv2[dst] = (__bf16)acc[sm][sn][j];
            }
}

// ---------------------------------------------------------------------------
// Stage 2: pooled projections. Per (sel,n): (1024 x 3136) @ (3136 x 64) + bias.
// 32x64 tile, k-tile 64, double-buffered async LDS. Grid (32, 36), block 256.
// ---------------------------------------------------------------------------
__global__ void __launch_bounds__(256) k_gemm_pool(const __bf16* __restrict__ qkv2,
                                                   const __bf16* __restrict__ wpT,
                                                   const float* __restrict__ bq,
                                                   const float* __restrict__ bk,
                                                   const float* __restrict__ bv,
                                                   float* __restrict__ pooled) {
    __shared__ __bf16 As[2][32][72];
    __shared__ __bf16 Bs[2][64][72];
    int by = blockIdx.y;                 // sel*12 + n
    int sel = by / 12, nn = by % 12;
    const __bf16* A  = qkv2 + (size_t)by * 1024 * 3136;
    const __bf16* BT = wpT  + (size_t)sel * 64 * 3136;
    const float* bias = (sel == 0) ? bq : (sel == 1 ? bk : bv);
    int tid = threadIdx.x, wave = tid >> 5, lane = tid & 31;
    int mi = wave & 1, ni = wave >> 1;   // 2 x 4 -> 16x16 per wave
    int row0 = blockIdx.x * 32;

    auto issue = [&](int buf, int kt) {
        {   // A: 32x64 -> 256 chunks, 1 per thread
            int r = tid >> 3, c8 = (tid & 7) * 8;
            async_copy_b128(lds_off(&As[buf][r][c8]),
                            A + (size_t)(row0 + r) * 3136 + kt + c8);
        }
#pragma unroll
        for (int q = 0; q < 2; ++q) {    // B: 64x64 -> 512 chunks, 2 per thread
            int cidx = tid * 2 + q;
            int r = cidx >> 3, c8 = (cidx & 7) * 8;
            async_copy_b128(lds_off(&Bs[buf][r][c8]),
                            BT + (size_t)r * 3136 + kt + c8);
        }
    };

    v8f acc = {};
    issue(0, 0);
    for (int it = 0; it < 49; ++it) {    // 3136 / 64
        int cur = it & 1;
        if (it + 1 < 49) {
            issue(cur ^ 1, (it + 1) * 64);
            WAIT_ASYNC(3);
        } else {
            WAIT_ASYNC(0);
        }
        __syncthreads();
#pragma unroll
        for (int ks = 0; ks < 64; ks += 32) {
            v16bf a = frag_a(&As[cur][mi * 16][ks], 72, lane);
            v16bf b = frag_b(&Bs[cur][ni * 16][ks], 72, lane);
            acc = wmma_bf16(a, b, acc);
        }
        __syncthreads();
    }

    int n = lane & 15, m0 = (lane >> 4) * 8;
    int d = ni * 16 + n;
    float bsv = bias[d];
    float* out = pooled + (size_t)sel * (16 * 12 * 64 * 64);
#pragma unroll
    for (int j = 0; j < 8; ++j) {
        int row = row0 + mi * 16 + m0 + j;
        int b_ = row >> 6, t = row & 63;
        out[(((size_t)b_ * 12 + nn) * 64 + t) * 64 + d] = acc[j] + bsv;
    }
}

// ---------------------------------------------------------------------------
// Layernorm over last dim (64), in place. One thread per row.
// ---------------------------------------------------------------------------
__global__ void k_layernorm(float* __restrict__ x, const float* __restrict__ g,
                            const float* __restrict__ be, int rows) {
    int r = blockIdx.x * blockDim.x + threadIdx.x;
    if (r >= rows) return;
    float* p = x + (size_t)r * 64;
    float m = 0.f;
#pragma unroll 8
    for (int i = 0; i < 64; ++i) m += p[i];
    m *= (1.f / 64.f);
    float v = 0.f;
#pragma unroll 8
    for (int i = 0; i < 64; ++i) { float d = p[i] - m; v += d * d; }
    v *= (1.f / 64.f);
    float rs = rsqrtf(v + 1e-5f);
#pragma unroll 8
    for (int i = 0; i < 64; ++i) p[i] = (p[i] - m) * rs * g[i] + be[i];
}

// ---------------------------------------------------------------------------
// Stage 3: attention per (b,n). Block = 128 threads (4 waves, 16 q-rows each).
// ---------------------------------------------------------------------------
__global__ void __launch_bounds__(128) k_attn(const float* __restrict__ pooled,
                                              const __bf16* __restrict__ relbf,
                                              float* __restrict__ attn_out) {
    __shared__ __bf16 qb[64][72];
    __shared__ __bf16 kb[64][72];      // reused for softmax probs later
    __shared__ __bf16 vT[64][72];      // v transposed: vT[c][s]
    __shared__ float  Sl[64][65];
    __shared__ float  Pl[64][128];
    int bn = blockIdx.x;               // b*12 + n
    int b_ = bn / 12, nn = bn % 12;
    const float* qg = pooled + (size_t)bn * 64 * 64;
    const float* kg = pooled + (size_t)12288 * 64 + (size_t)bn * 64 * 64;
    const float* vg = pooled + (size_t)2 * 12288 * 64 + (size_t)bn * 64 * 64;
    int tid = threadIdx.x, wave = tid >> 5, lane = tid & 31;

    for (int i = tid; i < 64 * 64; i += 128) {
        int t = i >> 6, c = i & 63;
        qb[t][c] = (__bf16)qg[i];
        kb[t][c] = (__bf16)kg[i];
        vT[c][t] = (__bf16)vg[i];
    }
    __syncthreads();

    int t0 = wave * 16;
    v8f zero = {};
    v8f sacc[4];
#pragma unroll
    for (int st = 0; st < 4; ++st) {
        sacc[st] = zero;
#pragma unroll
        for (int kk = 0; kk < 64; kk += 32) {
            v16bf a = frag_a(&qb[t0][kk], 72, lane);
            v16bf b = frag_b(&kb[st * 16][kk], 72, lane);
            sacc[st] = wmma_bf16(a, b, sacc[st]);
        }
    }
#pragma unroll
    for (int pt = 0; pt < 8; ++pt) {
        v8f pacc = zero;
#pragma unroll
        for (int kk = 0; kk < 64; kk += 32) {
            v16bf a = frag_a(&qb[t0][kk], 72, lane);
            v16bf b = frag_b(relbf + (size_t)(pt * 16) * 64 + kk, 64, lane);
            pacc = wmma_bf16(a, b, pacc);
        }
        int n = lane & 15, m0 = (lane >> 4) * 8;
#pragma unroll
        for (int j = 0; j < 8; ++j) Pl[t0 + m0 + j][pt * 16 + n] = pacc[j];
    }
    {
        const float scale = 0.125f;    // 64^-0.5
        int n = lane & 15, m0 = (lane >> 4) * 8;
#pragma unroll
        for (int st = 0; st < 4; ++st)
#pragma unroll
            for (int j = 0; j < 8; ++j) {
                int t = t0 + m0 + j, s = st * 16 + n;
                Sl[t][s] = scale * sacc[st][j] + Pl[t][t - s + 63];
            }
    }
    __syncthreads();
    if (tid < 64) {
        float mx = -3.4e38f;
        for (int s = 0; s < 64; ++s) mx = fmaxf(mx, Sl[tid][s]);
        float sum = 0.f;
        for (int s = 0; s < 64; ++s) { float e = __expf(Sl[tid][s] - mx); Sl[tid][s] = e; sum += e; }
        float inv = 1.f / sum;
        for (int s = 0; s < 64; ++s) kb[tid][s] = (__bf16)(Sl[tid][s] * inv);
    }
    __syncthreads();
#pragma unroll
    for (int ct = 0; ct < 4; ++ct) {
        v8f oacc = zero;
#pragma unroll
        for (int kk = 0; kk < 64; kk += 32) {
            v16bf a = frag_a(&kb[t0][kk], 72, lane);
            v16bf b = frag_b(&vT[ct * 16][kk], 72, lane);
            oacc = wmma_bf16(a, b, oacc);
        }
        int n = lane & 15, m0 = (lane >> 4) * 8;
#pragma unroll
        for (int j = 0; j < 8; ++j) {
            int t = t0 + m0 + j, c = ct * 16 + n;
            attn_out[((size_t)b_ * 64 + t) * 768 + nn * 64 + c] = oacc[j] + qg[t * 64 + c];
        }
    }
}

// ---------------------------------------------------------------------------
// Stage 4: out = attn_out @ W_proj + b_proj. M=1024, K=768, N=768.
// ---------------------------------------------------------------------------
__global__ void __launch_bounds__(256) k_gemm_proj(const float* __restrict__ Ain,
                                                   const __bf16* __restrict__ BT,
                                                   const float* __restrict__ bias,
                                                   float* __restrict__ out) {
    __shared__ __bf16 As[64][40];
    __shared__ __bf16 Bs[32][40];
    int tid = threadIdx.x, wave = tid >> 5, lane = tid & 31;
    int mi = wave & 3, ni = wave >> 2;
    int row0 = blockIdx.x * 64, col0 = blockIdx.y * 32;
    v8f acc = {};
    for (int kt = 0; kt < 768; kt += 32) {
        {   int r = tid >> 2, c = (tid & 3) * 8;
            const float* src = Ain + (size_t)(row0 + r) * 768 + kt + c;
#pragma unroll
            for (int j = 0; j < 8; ++j) As[r][c + j] = (__bf16)src[j];
        }
        if (tid < 128) {
            int r = tid >> 2, c = (tid & 3) * 8;
            const __bf16* src = BT + (size_t)(col0 + r) * 768 + kt + c;
#pragma unroll
            for (int j = 0; j < 8; ++j) Bs[r][c + j] = src[j];
        }
        __syncthreads();
        v16bf a = frag_a(&As[mi * 16][0], 40, lane);
        v16bf b = frag_b(&Bs[ni * 16][0], 40, lane);
        acc = wmma_bf16(a, b, acc);
        __syncthreads();
    }
    int n = lane & 15, m0 = (lane >> 4) * 8;
    int col = col0 + ni * 16 + n;
    float bv = bias[col];
#pragma unroll
    for (int j = 0; j < 8; ++j)
        out[(size_t)(row0 + mi * 16 + m0 + j) * 768 + col] = acc[j] + bv;
}

// ---------------------------------------------------------------------------
// Launcher
// ---------------------------------------------------------------------------
extern "C" void kernel_launch(void* const* d_in, const int* in_sizes, int n_in,
                              void* d_out, int out_size, void* d_ws, size_t ws_size,
                              hipStream_t stream) {
    const float* x      = (const float*)d_in[0];
    const float* W_qkv  = (const float*)d_in[1];
    const float* Wpq    = (const float*)d_in[2];
    const float* bpq    = (const float*)d_in[3];
    const float* Wpk    = (const float*)d_in[4];
    const float* bpk    = (const float*)d_in[5];
    const float* Wpv    = (const float*)d_in[6];
    const float* bpv    = (const float*)d_in[7];
    const float* g_q    = (const float*)d_in[8];
    const float* be_q   = (const float*)d_in[9];
    const float* g_k    = (const float*)d_in[10];
    const float* be_k   = (const float*)d_in[11];
    const float* g_v    = (const float*)d_in[12];
    const float* be_v   = (const float*)d_in[13];
    const float* relpos = (const float*)d_in[14];
    const float* W_proj = (const float*)d_in[15];
    const float* b_proj = (const float*)d_in[16];
    float* out = (float*)d_out;

    char* wsp = (char*)d_ws;
    auto carve = [&](size_t bytes) -> char* {
        char* p = wsp;
        wsp += (bytes + 255) & ~(size_t)255;
        return p;
    };
    const size_t NX = (size_t)16 * 64 * 49 * 768;
    __bf16* x_bf    = (__bf16*)carve(NX * 2);
    __bf16* wqkvT   = (__bf16*)carve((size_t)2304 * 768 * 2);
    __bf16* qkv2    = (__bf16*)carve((size_t)36 * 1024 * 3136 * 2);
    __bf16* wpT     = (__bf16*)carve((size_t)3 * 64 * 3136 * 2);
    float*  pooled  = (float*)carve((size_t)3 * 12288 * 64 * 4);
    __bf16* relbf   = (__bf16*)carve((size_t)128 * 64 * 2);
    float*  attn_o  = (float*)carve((size_t)1024 * 768 * 4);
    __bf16* wprojT  = (__bf16*)carve((size_t)768 * 768 * 2);

    k_f32_to_bf16<<<(unsigned)((NX + 255) / 256), 256, 0, stream>>>(x, x_bf, NX);
    k_transpose_bf16<<<(768 * 2304 + 255) / 256, 256, 0, stream>>>(W_qkv, wqkvT, 768, 2304);
    k_rearrange_wp<<<(64 * 3136 + 255) / 256, 256, 0, stream>>>(Wpq, wpT);
    k_rearrange_wp<<<(64 * 3136 + 255) / 256, 256, 0, stream>>>(Wpk, wpT + (size_t)64 * 3136);
    k_rearrange_wp<<<(64 * 3136 + 255) / 256, 256, 0, stream>>>(Wpv, wpT + (size_t)2 * 64 * 3136);
    k_relpos<<<(128 * 64 + 255) / 256, 256, 0, stream>>>(relpos, relbf);
    k_transpose_bf16<<<(768 * 768 + 255) / 256, 256, 0, stream>>>(W_proj, wprojT, 768, 768);

    k_gemm_qkv<<<dim3(392, 36), 256, 0, stream>>>(x_bf, wqkvT, qkv2);
    k_gemm_pool<<<dim3(32, 36), 256, 0, stream>>>(qkv2, wpT, bpq, bpk, bpv, pooled);
    k_layernorm<<<48, 256, 0, stream>>>(pooled, g_q, be_q, 12288);
    k_layernorm<<<48, 256, 0, stream>>>(pooled + (size_t)12288 * 64, g_k, be_k, 12288);
    k_layernorm<<<48, 256, 0, stream>>>(pooled + (size_t)2 * 12288 * 64, g_v, be_v, 12288);
    k_attn<<<192, 128, 0, stream>>>(pooled, relbf, attn_o);
    k_gemm_proj<<<dim3(16, 24), 256, 0, stream>>>(attn_o, wprojT, b_proj, out);
}
